// TransformerBlock_16389595202209
// MI455X (gfx1250) — compile-verified, matrix operand order checked by reading, not run on hardware
//
#include <hip/hip_runtime.h>
#include <hip/hip_bf16.h>

// ---------------------------------------------------------------------------
// Transformer block for MI455X (gfx1250, wave32, WMMA).
// Precision strategy: bf16 WMMA (v_wmma_f32_16x16x32_bf16) everywhere, fp32
// accumulate. Dominant cost is the mandatory 1.07 GB attn-prob output write,
// so attention uses flash-style online softmax + a recompute pass that emits
// normalized probabilities with non-temporal stores. V tiles are staged into
// LDS with gfx1250 async-to-LDS DMA (ASYNCcnt) overlapped with score WMMAs.
// ---------------------------------------------------------------------------

#define D_MODEL 1024
#define N_HEADS 16
#define DK      64
#define D_FF    4096
#define SEQ_T   4096
#define QKV_N   (3 * D_MODEL)

typedef __attribute__((ext_vector_type(16))) __bf16 bf16x16;
typedef __attribute__((ext_vector_type(8)))  float  floatx8;

union FragB { bf16x16 v; unsigned short u[16]; uint4 q[2]; };

__device__ __forceinline__ unsigned short f2bf(float f) {
    unsigned int x = __float_as_uint(f);
    x += 0x7FFFu + ((x >> 16) & 1u);          // round-to-nearest-even
    return (unsigned short)(x >> 16);
}

// 16-bit A/B fragment (16x32, K-major chunks): lane holds 8 contiguous halves
// at K = koff and 8 at K = koff+16, koff = 8*(lane>>4).  (ISA 7.12.2)
__device__ __forceinline__ bf16x16 gfrag(const unsigned short* __restrict__ base,
                                         int row, int stride, int kbase, int lane) {
    const unsigned short* p = base + (size_t)row * stride + kbase + ((lane >> 4) << 3);
    FragB f;
    f.q[0] = *(const uint4*)(p);
    f.q[1] = *(const uint4*)(p + 16);
    return f.v;
}

__device__ __forceinline__ floatx8 wmma_bf16(bf16x16 a, bf16x16 b, floatx8 c) {
    return __builtin_amdgcn_wmma_f32_16x16x32_bf16(false, a, false, b,
                                                   (short)0, c, false, false);
}

// A-fragment of the 16x32 P tile held in LDS (row stride 32 halves).
__device__ __forceinline__ bf16x16 lds_afrag(const unsigned short* sm, int lane) {
    const unsigned short* p = sm + (lane & 15) * 32 + ((lane >> 4) << 3);
    FragB f;
    f.q[0] = *(const uint4*)(p);
    f.q[1] = *(const uint4*)(p + 16);
    return f.v;
}

// B-fragment of V (32 keys x 16 dk-cols, chunk c) gathered from LDS.
__device__ __forceinline__ bf16x16 lds_vfrag(const unsigned short* vt, int c, int lane) {
    int n    = (lane & 15) + c * 16;
    int koff = (lane >> 4) << 3;
    FragB f;
#pragma unroll
    for (int e = 0; e < 8; ++e) f.u[e]     = vt[(koff + e) * 64 + n];
#pragma unroll
    for (int e = 0; e < 8; ++e) f.u[8 + e] = vt[(koff + 16 + e) * 64 + n];
    return f.v;
}

// ---------------------------------------------------------------------------
__global__ void cast_bf16_k(const float* __restrict__ in,
                            unsigned short* __restrict__ out, int n4) {
    int i = blockIdx.x * 256 + threadIdx.x;
    if (i >= n4) return;
    float4 v = ((const float4*)in)[i];
    uint2 pk;
    pk.x = (unsigned)f2bf(v.x) | ((unsigned)f2bf(v.y) << 16);
    pk.y = (unsigned)f2bf(v.z) | ((unsigned)f2bf(v.w) << 16);
    ((uint2*)out)[i] = pk;
}

// Fused LayerNorm (row = 1024) + bf16 cast. One row per 256-thread block.
__global__ __launch_bounds__(256) void ln_cast_k(const float* __restrict__ x,
                                                 const float* __restrict__ g,
                                                 const float* __restrict__ b,
                                                 unsigned short* __restrict__ out) {
    int row = blockIdx.x, t = threadIdx.x;
    float4 v = ((const float4*)(x + (size_t)row * D_MODEL))[t];
    float s  = v.x + v.y + v.z + v.w;
    float s2 = v.x * v.x + v.y * v.y + v.z * v.z + v.w * v.w;
#pragma unroll
    for (int off = 16; off >= 1; off >>= 1) {
        s  += __shfl_xor(s,  off, 32);
        s2 += __shfl_xor(s2, off, 32);
    }
    __shared__ float rs[8], rs2[8];
    if ((t & 31) == 0) { rs[t >> 5] = s; rs2[t >> 5] = s2; }
    __syncthreads();
    float ts = 0.f, ts2 = 0.f;
#pragma unroll
    for (int i = 0; i < 8; ++i) { ts += rs[i]; ts2 += rs2[i]; }
    float mu  = ts  * (1.0f / D_MODEL);
    float var = ts2 * (1.0f / D_MODEL) - mu * mu;
    float inv = rsqrtf(var + 1e-5f);
    float4 gv = ((const float4*)g)[t];
    float4 bv = ((const float4*)b)[t];
    float y0 = (v.x - mu) * inv * gv.x + bv.x;
    float y1 = (v.y - mu) * inv * gv.y + bv.y;
    float y2 = (v.z - mu) * inv * gv.z + bv.z;
    float y3 = (v.w - mu) * inv * gv.w + bv.w;
    uint2 pk;
    pk.x = (unsigned)f2bf(y0) | ((unsigned)f2bf(y1) << 16);
    pk.y = (unsigned)f2bf(y2) | ((unsigned)f2bf(y3) << 16);
    ((uint2*)(out + (size_t)row * D_MODEL))[t] = pk;
}

// ---------------------------------------------------------------------------
// C = A[MxK](bf16) * W[NxK]^T(bf16) + bias, tiled 64(M)x128(N) per block,
// 16x64 per wave; 4 WMMAs share one A fragment per K-step of 32.
// MODE 0: store bf16, scale cols < D_MODEL by 1/sqrt(dk) (QKV, folds q-scale)
// MODE 1: exact GELU, store bf16
// MODE 2: + residual, store fp32
template <int MODE>
__global__ __launch_bounds__(256) void gemm_bf16_k(
    const unsigned short* __restrict__ A, const unsigned short* __restrict__ W,
    const float* __restrict__ bias, const float* __restrict__ resid,
    void* __restrict__ outp, int N, int K) {
    int lane = threadIdx.x & 31;
    int wid  = threadIdx.x >> 5;
    int m0 = blockIdx.y * 64  + (wid & 3) * 16;
    int n0 = blockIdx.x * 128 + (wid >> 2) * 64;

    floatx8 acc[4] = {};
    int ar = m0 + (lane & 15);
    int nr = n0 + (lane & 15);
    for (int k = 0; k < K; k += 32) {
        bf16x16 a = gfrag(A, ar, K, k, lane);
#pragma unroll
        for (int c = 0; c < 4; ++c) {
            bf16x16 b = gfrag(W, nr + c * 16, K, k, lane);
            acc[c] = wmma_bf16(a, b, acc[c]);
        }
    }
    int colb = n0 + (lane & 15);
    int rowb = m0 + ((lane >> 4) << 3);
#pragma unroll
    for (int c = 0; c < 4; ++c) {
        int col  = colb + c * 16;
        float bs = bias[col];
#pragma unroll
        for (int r = 0; r < 8; ++r) {
            int row = rowb + r;
            float v = acc[c][r] + bs;
            if (MODE == 0) {
                if (col < D_MODEL) v *= 0.125f;   // 1/sqrt(64) on Q
                ((unsigned short*)outp)[(size_t)row * N + col] = f2bf(v);
            } else if (MODE == 1) {
                v = 0.5f * v * (1.0f + erff(v * 0.70710678118654752f));
                ((unsigned short*)outp)[(size_t)row * N + col] = f2bf(v);
            } else {
                ((float*)outp)[(size_t)row * N + col] =
                    v + resid[(size_t)row * N + col];
            }
        }
    }
}

// ---------------------------------------------------------------------------
// Flash-style causal attention, one wave per (head, 16-query tile).
// Pass 1: online softmax + O accumulation (8 WMMAs per 32-key pair); the V
//   tile is DMA'd into LDS with global_load_async_to_lds_b128 (ASYNCcnt),
//   overlapping the score WMMAs; s_wait_asynccnt 0 gates the P*V WMMAs.
// Pass 2: recompute scores, stream exp(s-m)/l (and causal zeros) non-temporal.
__global__ __launch_bounds__(32) void attn_k(const unsigned short* __restrict__ qkv,
                                             unsigned short* __restrict__ attn_out,
                                             float* __restrict__ probs) {
    __shared__ unsigned short p_tile[16 * 32];
    __shared__ unsigned short v_tile[32 * 64];
    int lane = threadIdx.x;
    int q0 = blockIdx.x * 16;
    int h  = blockIdx.y;
    const unsigned short* qp = qkv + h * DK;                 // Q (pre-scaled)
    const unsigned short* kp = qkv + D_MODEL + h * DK;       // K
    const unsigned short* vp = qkv + 2 * D_MODEL + h * DK;   // V
    int l15 = lane & 15;
    int lhi = (lane >> 4) << 3;                              // 8*(lane>=16)

    bf16x16 aq0 = gfrag(qp, q0 + l15, QKV_N, 0,  lane);
    bf16x16 aq1 = gfrag(qp, q0 + l15, QKV_N, 32, lane);

    // low 32 bits of a shared-aperture generic pointer == LDS byte offset
    unsigned vlds = (unsigned)(size_t)(&v_tile[lane * 64]);

    float m[8], lsum[8];
    floatx8 o[4] = {};
#pragma unroll
    for (int r = 0; r < 8; ++r) { m[r] = -1e30f; lsum[r] = 0.f; }

    int npair = (q0 + 16 + 31) >> 5;   // causal 32-key pairs needed
    for (int pb = 0; pb < npair; ++pb) {
        int k0 = pb << 5;
        // async DMA: V tile (32 keys x 64 bf16) -> LDS, one row per lane.
        // offset:N is added to BOTH the LDS and the global address.
        {
            const unsigned short* gsrc = vp + (size_t)(k0 + lane) * QKV_N;
#define ASYNC_V_LD(OFF)                                                       \
            asm volatile("global_load_async_to_lds_b128 %0, %1, off offset:" #OFF \
                         :: "v"(vlds), "v"(gsrc) : "memory")
            ASYNC_V_LD(0);  ASYNC_V_LD(16); ASYNC_V_LD(32);  ASYNC_V_LD(48);
            ASYNC_V_LD(64); ASYNC_V_LD(80); ASYNC_V_LD(96);  ASYNC_V_LD(112);
#undef ASYNC_V_LD
        }
        // scores: two 16x16 tiles, dk=64 -> 2 WMMAs each (overlaps the DMA)
        floatx8 s0 = {}, s1 = {};
        s0 = wmma_bf16(aq0, gfrag(kp, k0 + l15,      QKV_N, 0,  lane), s0);
        s0 = wmma_bf16(aq1, gfrag(kp, k0 + l15,      QKV_N, 32, lane), s0);
        s1 = wmma_bf16(aq0, gfrag(kp, k0 + 16 + l15, QKV_N, 0,  lane), s1);
        s1 = wmma_bf16(aq1, gfrag(kp, k0 + 16 + l15, QKV_N, 32, lane), s1);

        int ck = k0 + l15;
#pragma unroll
        for (int r = 0; r < 8; ++r) {
            int qrow = q0 + r + lhi;
            float a0 = (ck      <= qrow) ? s0[r] : -1e30f;   // causal mask
            float a1 = (ck + 16 <= qrow) ? s1[r] : -1e30f;
            float mx = fmaxf(a0, a1);
            mx = fmaxf(mx, __shfl_xor(mx, 1, 32));
            mx = fmaxf(mx, __shfl_xor(mx, 2, 32));
            mx = fmaxf(mx, __shfl_xor(mx, 4, 32));
            mx = fmaxf(mx, __shfl_xor(mx, 8, 32));
            float mn   = fmaxf(m[r], mx);
            float corr = __expf(m[r] - mn);
            float p0 = __expf(a0 - mn);
            float p1 = __expf(a1 - mn);
            float rsu = p0 + p1;
            rsu += __shfl_xor(rsu, 1, 32);
            rsu += __shfl_xor(rsu, 2, 32);
            rsu += __shfl_xor(rsu, 4, 32);
            rsu += __shfl_xor(rsu, 8, 32);
            m[r]    = mn;
            lsum[r] = lsum[r] * corr + rsu;
            o[0][r] *= corr; o[1][r] *= corr; o[2][r] *= corr; o[3][r] *= corr;
            int rr = r + lhi;
            p_tile[rr * 32 + l15]      = f2bf(p0);
            p_tile[rr * 32 + 16 + l15] = f2bf(p1);
        }
        // V DMA must have landed before the P*V WMMAs read v_tile
        asm volatile("s_wait_asynccnt 0x0" ::: "memory");
        __syncthreads();   // LDS C-layout -> A-layout transpose point
        bf16x16 ap = lds_afrag(p_tile, lane);
#pragma unroll
        for (int c = 0; c < 4; ++c)
            o[c] = wmma_bf16(ap, lds_vfrag(v_tile, c, lane), o[c]);
        __syncthreads();
    }

    float inv[8];
#pragma unroll
    for (int r = 0; r < 8; ++r) inv[r] = 1.0f / lsum[r];
#pragma unroll
    for (int c = 0; c < 4; ++c)
#pragma unroll
        for (int r = 0; r < 8; ++r)
            attn_out[(size_t)(q0 + r + lhi) * D_MODEL + h * DK + c * 16 + l15] =
                f2bf(o[c][r] * inv[r]);

    // pass 2: normalized probabilities + causal zero-fill, non-temporal
    float* po = probs + ((size_t)h * SEQ_T + q0) * SEQ_T;
    for (int pb = 0; pb < SEQ_T / 32; ++pb) {
        int k0 = pb << 5;
        if (pb < npair) {
            floatx8 s0 = {}, s1 = {};
            s0 = wmma_bf16(aq0, gfrag(kp, k0 + l15,      QKV_N, 0,  lane), s0);
            s0 = wmma_bf16(aq1, gfrag(kp, k0 + l15,      QKV_N, 32, lane), s0);
            s1 = wmma_bf16(aq0, gfrag(kp, k0 + 16 + l15, QKV_N, 0,  lane), s1);
            s1 = wmma_bf16(aq1, gfrag(kp, k0 + 16 + l15, QKV_N, 32, lane), s1);
            int ck = k0 + l15;
#pragma unroll
            for (int r = 0; r < 8; ++r) {
                int rr = r + lhi;
                int qrow = q0 + rr;
                float a0 = (ck      <= qrow) ? s0[r] : -1e30f;
                float a1 = (ck + 16 <= qrow) ? s1[r] : -1e30f;
                float w0 = __expf(a0 - m[r]) * inv[r];
                float w1 = __expf(a1 - m[r]) * inv[r];
                size_t idx = (size_t)rr * SEQ_T + k0 + l15;
                __builtin_nontemporal_store(w0, po + idx);
                __builtin_nontemporal_store(w1, po + idx + 16);
            }
        } else {
#pragma unroll
            for (int r = 0; r < 8; ++r) {
                size_t idx = (size_t)(r + lhi) * SEQ_T + k0 + l15;
                __builtin_nontemporal_store(0.0f, po + idx);
                __builtin_nontemporal_store(0.0f, po + idx + 16);
            }
        }
    }
}

// ---------------------------------------------------------------------------
extern "C" void kernel_launch(void* const* d_in, const int* in_sizes, int n_in,
                              void* d_out, int out_size, void* d_ws, size_t ws_size,
                              hipStream_t stream) {
    const float* x     = (const float*)d_in[0];
    const float* ln1_g = (const float*)d_in[1];
    const float* ln1_b = (const float*)d_in[2];
    const float* w_qkv = (const float*)d_in[3];
    const float* b_qkv = (const float*)d_in[4];
    const float* w_out = (const float*)d_in[5];
    const float* b_out = (const float*)d_in[6];
    const float* ln2_g = (const float*)d_in[7];
    const float* ln2_b = (const float*)d_in[8];
    const float* w1    = (const float*)d_in[9];
    const float* b1    = (const float*)d_in[10];
    const float* w2    = (const float*)d_in[11];
    const float* b2    = (const float*)d_in[12];
    (void)in_sizes; (void)n_in; (void)out_size; (void)ws_size;

    float* out_x    = (float*)d_out;                       // [4096,1024]
    float* out_attn = out_x + (size_t)SEQ_T * D_MODEL;     // [16,4096,4096]

    char* ws = (char*)d_ws;
    size_t off = 0;
    auto carve = [&](size_t bytes) -> void* {
        void* p = ws + off;
        off = (off + bytes + 255) & ~(size_t)255;
        return p;
    };
    unsigned short* wqkv_b = (unsigned short*)carve((size_t)QKV_N * D_MODEL * 2);
    unsigned short* wout_b = (unsigned short*)carve((size_t)D_MODEL * D_MODEL * 2);
    unsigned short* w1_b   = (unsigned short*)carve((size_t)D_FF * D_MODEL * 2);
    unsigned short* w2_b   = (unsigned short*)carve((size_t)D_MODEL * D_FF * 2);
    unsigned short* h_b    = (unsigned short*)carve((size_t)SEQ_T * D_MODEL * 2);
    unsigned short* qkv_b  = (unsigned short*)carve((size_t)SEQ_T * QKV_N * 2);
    unsigned short* attn_b = (unsigned short*)carve((size_t)SEQ_T * D_MODEL * 2);
    float*          x1     = (float*)carve((size_t)SEQ_T * D_MODEL * 4);
    unsigned short* ff_b   = (unsigned short*)carve((size_t)SEQ_T * D_FF * 2);

    // weights -> bf16
    cast_bf16_k<<<(QKV_N * D_MODEL / 4 + 255) / 256, 256, 0, stream>>>(w_qkv, wqkv_b, QKV_N * D_MODEL / 4);
    cast_bf16_k<<<(D_MODEL * D_MODEL / 4 + 255) / 256, 256, 0, stream>>>(w_out, wout_b, D_MODEL * D_MODEL / 4);
    cast_bf16_k<<<(D_FF * D_MODEL / 4 + 255) / 256, 256, 0, stream>>>(w1, w1_b, D_FF * D_MODEL / 4);
    cast_bf16_k<<<(D_MODEL * D_FF / 4 + 255) / 256, 256, 0, stream>>>(w2, w2_b, D_MODEL * D_FF / 4);

    // h1 = LN1(x) -> bf16
    ln_cast_k<<<SEQ_T, 256, 0, stream>>>(x, ln1_g, ln1_b, h_b);
    // qkv = h1 @ w_qkv^T + b_qkv (Q pre-scaled by 1/8) -> bf16
    gemm_bf16_k<0><<<dim3(QKV_N / 128, SEQ_T / 64), 256, 0, stream>>>(
        h_b, wqkv_b, b_qkv, nullptr, qkv_b, QKV_N, D_MODEL);
    // attention -> attn_b (bf16) + probabilities -> d_out
    attn_k<<<dim3(SEQ_T / 16, N_HEADS), 32, 0, stream>>>(qkv_b, attn_b, out_attn);
    // x1 = x + attn @ w_out^T + b_out (fp32)
    gemm_bf16_k<2><<<dim3(D_MODEL / 128, SEQ_T / 64), 256, 0, stream>>>(
        attn_b, wout_b, b_out, x, x1, D_MODEL, D_MODEL);
    // h2 = LN2(x1) -> bf16
    ln_cast_k<<<SEQ_T, 256, 0, stream>>>(x1, ln2_g, ln2_b, h_b);
    // ff = gelu(h2 @ w1^T + b1) -> bf16
    gemm_bf16_k<1><<<dim3(D_FF / 128, SEQ_T / 64), 256, 0, stream>>>(
        h_b, w1_b, b1, nullptr, ff_b, D_FF, D_MODEL);
    // out_x = x1 + ff @ w2^T + b2 (fp32)
    gemm_bf16_k<2><<<dim3(D_MODEL / 128, SEQ_T / 64), 256, 0, stream>>>(
        ff_b, w2_b, b2, x1, out_x, D_MODEL, D_FF);
}
